// MultiHeadAttention_59742995087482
// MI455X (gfx1250) — matrix operand order repaired and checked by reading, not compiled, not run on hardware
//
#include <hip/hip_runtime.h>
#include <hip/hip_bf16.h>
#include <stdint.h>

// ---------------------------------------------------------------------------
// MHA on gfx1250 (MI455X): bf16 WMMA everywhere, flash-attention inner loop.
//   S=2048, E=1024, H=16, dk=dv=64.  All matmuls via v_wmma_f32_16x16x32_bf16.
//   Softmax runs in base-2 (scale*log2e folded into Q projection epilogue);
//   row reductions use ds_swizzle xor butterflies (ISA §11 SWAPX patterns).
// ---------------------------------------------------------------------------

#define S_LEN 2048
#define E_DIM 1024
#define HD    1024      // H * DK == H * DV
#define NHEAD 16
#define DK    64
#define DV    64

typedef __attribute__((ext_vector_type(16))) __bf16   v16bf;
typedef __attribute__((ext_vector_type(8)))  float    v8f;
typedef __attribute__((ext_vector_type(4)))  uint32_t v4u;

__device__ __forceinline__ uint16_t f2bf(float f) {
  uint32_t u = __float_as_uint(f);
  u += 0x7fffu + ((u >> 16) & 1u);     // round-to-nearest-even
  return (uint16_t)(u >> 16);
}

__device__ __forceinline__ v16bf make_frag(v4u lo, v4u hi) {
  union { v4u u[2]; v16bf f; } t;
  t.u[0] = lo; t.u[1] = hi;
  return t.f;
}
// contiguous 32B fragment (16B-aligned is enough -> two ds/global b128 loads)
__device__ __forceinline__ v16bf load_frag(const uint16_t* p) {
  return make_frag(*(const v4u*)p, *(const v4u*)(p + 8));
}
// A-matrix fragment from two 16B K chunks (callers pass adjacent chunks)
__device__ __forceinline__ v16bf load_frag2(const uint16_t* p0, const uint16_t* p1) {
  return make_frag(*(const v4u*)p0, *(const v4u*)p1);
}

// ds_swizzle xor butterfly within 16-lane row groups (SWAPX1/2/4/8)
template <int MASK>
__device__ __forceinline__ float swz_xor(float x) {
  return __int_as_float(
      __builtin_amdgcn_ds_swizzle(__float_as_int(x), (MASK << 10) | 0x1F));
}
__device__ __forceinline__ float rowmax16(float x) {
  x = fmaxf(x, swz_xor<1>(x));
  x = fmaxf(x, swz_xor<2>(x));
  x = fmaxf(x, swz_xor<4>(x));
  x = fmaxf(x, swz_xor<8>(x));
  return x;
}
__device__ __forceinline__ float rowsum16(float x) {
  x += swz_xor<1>(x);
  x += swz_xor<2>(x);
  x += swz_xor<4>(x);
  x += swz_xor<8>(x);
  return x;
}

// ---------------------------------------------------------------------------
// fp32 -> bf16 conversion (grid-stride)
// ---------------------------------------------------------------------------
__global__ void cvt_f32_bf16(const float* __restrict__ in,
                             uint16_t* __restrict__ out, int n) {
  int i = blockIdx.x * blockDim.x + threadIdx.x;
  int stride = gridDim.x * blockDim.x;
  for (; i < n; i += stride) out[i] = f2bf(in[i]);
}

// ---------------------------------------------------------------------------
// C[M,N] = alpha * A[M,K] * B[N,K]^T   (A,B bf16 row-major; C f32 or bf16)
// Block: 256 threads = 8 waves. Wave -> 16x64 strip, block tile 128x64.
// Per k-step (32): 1 A-frag, 4 B-frags, 4 WMMAs.
// ---------------------------------------------------------------------------
template <bool OUT_BF16>
__global__ __launch_bounds__(256)
void gemm_bf16(const uint16_t* __restrict__ A,
               const uint16_t* __restrict__ B,
               void* __restrict__ C,
               int M, int N, int K, float alpha) {
  const int lane = threadIdx.x & 31;
  const int wave = threadIdx.x >> 5;
  const int half = lane >> 4;
  const int l16  = lane & 15;

  const int mrow = blockIdx.y * 128 + wave * 16;
  const int ncol = blockIdx.x * 64;

  v8f zero = {};
  v8f acc[4];
#pragma unroll
  for (int t = 0; t < 4; ++t) acc[t] = zero;

  const uint16_t* aRow = A + (size_t)(mrow + l16) * K;

  for (int k0 = 0; k0 < K; k0 += 32) {
    if (k0 + 32 < K) __builtin_prefetch(aRow + k0 + 32, 0, 1);  // global_prefetch
    // A fragment (16x32 MxK): half 0 -> K 0..15, half 1 -> 16..31 (contiguous)
    v16bf af = load_frag2(aRow + k0 + 16 * half, aRow + k0 + 16 * half + 8);
#pragma unroll
    for (int t = 0; t < 4; ++t) {
      // B fragment (32x16 KxN): col = l16, K = 16*half..16*half+15 (contiguous)
      const uint16_t* bCol = B + (size_t)(ncol + t * 16 + l16) * K + k0 + 16 * half;
      v16bf bf = load_frag(bCol);
      acc[t] = __builtin_amdgcn_wmma_f32_16x16x32_bf16(
          false, af, false, bf, (short)0, acc[t], false, false);
    }
  }

#pragma unroll
  for (int t = 0; t < 4; ++t) {
#pragma unroll
    for (int v = 0; v < 8; ++v) {
      int r = mrow + v + 8 * half;        // C/D layout: VGPR v, lane half
      int c = ncol + t * 16 + l16;
      float val = acc[t][v] * alpha;
      if (OUT_BF16) ((uint16_t*)C)[(size_t)r * N + c] = f2bf(val);
      else          ((float*)C)[(size_t)r * N + c]    = val;
    }
  }
}

// ---------------------------------------------------------------------------
// Flash attention: block = 128 threads (4 waves), grid = (S/64, H).
// Each wave owns a 16-row query block of head blockIdx.y.
// Key tile = 64: 8 score WMMAs + base-2 online softmax + 8 PV WMMAs per tile.
// V tile staged in LDS transposed [dv][key] (stride-padded); P re-laid-out
// from C/D format to A-matrix format via per-wave LDS scratch.
// Q was pre-scaled by 0.125*log2(e) in the projection, so probs use exp2.
// ---------------------------------------------------------------------------
#define KT      64
#define VSTRIDE 72   // padded row stride (elems): 144B, 16B-aligned, bank-spread
#define PSTRIDE 72

__global__ __launch_bounds__(128)
void attention(const uint16_t* __restrict__ Q,    // [S, HD] bf16 (pre-scaled)
               const uint16_t* __restrict__ Kc,   // [S, HD] bf16
               const uint16_t* __restrict__ V,    // [S, HD] bf16
               uint16_t* __restrict__ A1) {       // [S, HD] bf16
  __shared__ uint16_t ldsV[KT * VSTRIDE];          // transposed V tile: [dv][key]
  __shared__ uint16_t ldsP[4 * 16 * PSTRIDE];      // per-wave P scratch

  const int lane = threadIdx.x & 31;
  const int wave = threadIdx.x >> 5;
  const int half = lane >> 4;
  const int l16  = lane & 15;

  const int h     = blockIdx.y;
  const int qbase = blockIdx.x * 64 + wave * 16;

  // Q fragments: 16x64 -> two 16x32 A-matrix fragments (loaded once)
  const uint16_t* qRow = Q + (size_t)(qbase + l16) * HD + h * DK;
  v16bf aq[2];
#pragma unroll
  for (int kk = 0; kk < 2; ++kk)
    aq[kk] = load_frag2(qRow + kk * 32 + 16 * half, qRow + kk * 32 + 16 * half + 8);

  float m_run[8], l_run[8];
#pragma unroll
  for (int v = 0; v < 8; ++v) { m_run[v] = -3.0e38f; l_run[v] = 0.0f; }

  v8f zero = {};
  v8f acc[4];   // O accumulator: 16 rows x 64 dv (4 tiles)
#pragma unroll
  for (int t = 0; t < 4; ++t) acc[t] = zero;

  uint16_t* myP = ldsP + wave * (16 * PSTRIDE);

  for (int kt = 0; kt < S_LEN / KT; ++kt) {
    const int kbase = kt * KT;

    __syncthreads();   // previous iteration's ldsV reads are done
    // cooperative transposed stage of V tile: ldsV[dv*VSTRIDE + key]
    for (int idx = threadIdx.x; idx < KT * 64; idx += 128) {
      int key = idx >> 6;
      int dv  = idx & 63;
      ldsV[dv * VSTRIDE + key] = V[(size_t)(kbase + key) * HD + h * DV + dv];
    }
    __syncthreads();

    // ---- scores: S = Q_h (16x64) . K_h^T -> 16x64 keys, four 16x16 tiles ----
    v8f sc[4];
#pragma unroll
    for (int t = 0; t < 4; ++t) sc[t] = zero;
#pragma unroll
    for (int kk = 0; kk < 2; ++kk) {
#pragma unroll
      for (int t = 0; t < 4; ++t) {
        const uint16_t* b =
            Kc + (size_t)(kbase + t * 16 + l16) * HD + h * DK + kk * 32 + 16 * half;
        sc[t] = __builtin_amdgcn_wmma_f32_16x16x32_bf16(
            false, aq[kk], false, load_frag(b), (short)0, sc[t], false, false);
      }
    }

    // ---- base-2 online softmax (scale already folded into Q) ----
#pragma unroll
    for (int v = 0; v < 8; ++v) {
      float s0 = sc[0][v], s1 = sc[1][v], s2 = sc[2][v], s3 = sc[3][v];
      float tmax = rowmax16(fmaxf(fmaxf(s0, s1), fmaxf(s2, s3)));
      float mnew = fmaxf(m_run[v], tmax);
      float corr = __builtin_amdgcn_exp2f(m_run[v] - mnew);
      float p0 = __builtin_amdgcn_exp2f(s0 - mnew);
      float p1 = __builtin_amdgcn_exp2f(s1 - mnew);
      float p2 = __builtin_amdgcn_exp2f(s2 - mnew);
      float p3 = __builtin_amdgcn_exp2f(s3 - mnew);
      float rsum = rowsum16((p0 + p1) + (p2 + p3));
      l_run[v] = l_run[v] * corr + rsum;
      m_run[v] = mnew;
#pragma unroll
      for (int t = 0; t < 4; ++t) acc[t][v] *= corr;
      // spill P (C/D layout) to per-wave LDS as bf16, rows padded to PSTRIDE
      uint16_t* pr = myP + (v + 8 * half) * PSTRIDE;
      pr[l16]      = f2bf(p0);
      pr[16 + l16] = f2bf(p1);
      pr[32 + l16] = f2bf(p2);
      pr[48 + l16] = f2bf(p3);
    }

    // ---- O += P (16x64) . V_tile (64x64) ----
    {
      const uint16_t* pRow = myP + l16 * PSTRIDE;
      v16bf pa[2];
#pragma unroll
      for (int kk = 0; kk < 2; ++kk)
        pa[kk] = load_frag2(pRow + kk * 32 + 16 * half,
                            pRow + kk * 32 + 16 * half + 8);
#pragma unroll
      for (int kk = 0; kk < 2; ++kk) {
#pragma unroll
        for (int t = 0; t < 4; ++t) {
          const uint16_t* bv =
              ldsV + (size_t)(t * 16 + l16) * VSTRIDE + kk * 32 + 16 * half;
          acc[t] = __builtin_amdgcn_wmma_f32_16x16x32_bf16(
              false, pa[kk], false, load_frag(bv), (short)0, acc[t], false, false);
        }
      }
    }
  }

  // ---- normalize and write A1 (concat heads on last dim), bf16 ----
#pragma unroll
  for (int v = 0; v < 8; ++v) {
    float rl = 1.0f / l_run[v];
    int r = qbase + v + 8 * half;
#pragma unroll
    for (int t = 0; t < 4; ++t)
      A1[(size_t)r * HD + h * DV + t * 16 + l16] = f2bf(acc[t][v] * rl);
  }
}

// ---------------------------------------------------------------------------
// Host-side launch
// ---------------------------------------------------------------------------
extern "C" void kernel_launch(void* const* d_in, const int* in_sizes, int n_in,
                              void* d_out, int out_size, void* d_ws, size_t ws_size,
                              hipStream_t stream) {
  (void)in_sizes; (void)n_in; (void)out_size; (void)ws_size;
  const float* x  = (const float*)d_in[0];
  const float* Wq = (const float*)d_in[1];
  const float* Wk = (const float*)d_in[2];
  const float* Wv = (const float*)d_in[3];
  const float* Wo = (const float*)d_in[4];

  char* ws = (char*)d_ws;
  size_t off = 0;
  auto carve = [&](size_t bytes) -> char* {
    char* p = ws + off;
    off += (bytes + 255) & ~(size_t)255;
    return p;
  };
  uint16_t* xb  = (uint16_t*)carve((size_t)S_LEN * E_DIM * 2);
  uint16_t* Wqb = (uint16_t*)carve((size_t)HD * E_DIM * 2);
  uint16_t* Wkb = (uint16_t*)carve((size_t)HD * E_DIM * 2);
  uint16_t* Wvb = (uint16_t*)carve((size_t)HD * E_DIM * 2);
  uint16_t* Wob = (uint16_t*)carve((size_t)HD * HD * 2);
  uint16_t* Qb  = (uint16_t*)carve((size_t)S_LEN * HD * 2);
  uint16_t* Kb  = (uint16_t*)carve((size_t)S_LEN * HD * 2);
  uint16_t* Vb  = (uint16_t*)carve((size_t)S_LEN * HD * 2);
  uint16_t* A1b = (uint16_t*)carve((size_t)S_LEN * HD * 2);

  // 1) convert everything to bf16
  cvt_f32_bf16<<<512, 256, 0, stream>>>(x,  xb,  S_LEN * E_DIM);
  cvt_f32_bf16<<<512, 256, 0, stream>>>(Wq, Wqb, HD * E_DIM);
  cvt_f32_bf16<<<512, 256, 0, stream>>>(Wk, Wkb, HD * E_DIM);
  cvt_f32_bf16<<<512, 256, 0, stream>>>(Wv, Wvb, HD * E_DIM);
  cvt_f32_bf16<<<512, 256, 0, stream>>>(Wo, Wob, HD * HD);

  // 2) QKV projections: [S,E] x [HD,E]^T -> [S,HD]  (bf16 out)
  //    Q gets alpha = (1/sqrt(dk)) * log2(e) so softmax can run in base-2.
  const float alphaQ = 0.125f * 1.44269504088896f;
  dim3 gGemm(HD / 64, S_LEN / 128);
  gemm_bf16<true><<<gGemm, 256, 0, stream>>>(xb, Wqb, Qb, S_LEN, HD, E_DIM, alphaQ);
  gemm_bf16<true><<<gGemm, 256, 0, stream>>>(xb, Wkb, Kb, S_LEN, HD, E_DIM, 1.0f);
  gemm_bf16<true><<<gGemm, 256, 0, stream>>>(xb, Wvb, Vb, S_LEN, HD, E_DIM, 1.0f);

  // 3) flash attention -> A1 [S,HD] bf16
  dim3 gAtt(S_LEN / 64, NHEAD);
  attention<<<gAtt, 128, 0, stream>>>(Qb, Kb, Vb, A1b);

  // 4) output projection: [S,HD] x [HD,HD]^T -> f32 d_out
  gemm_bf16<false><<<gGemm, 256, 0, stream>>>(A1b, Wob, (float*)d_out, S_LEN, HD, HD, 1.0f);
}
